// MaskedAttention_3813930958872
// MI455X (gfx1250) — matrix operand order repaired and checked by reading, not compiled, not run on hardware
//
#include <hip/hip_runtime.h>
#include <hip/hip_bf16.h>

// Problem constants (from reference): B=16, S=1024, D=768, H=12, HD=64
static constexpr int CB  = 16;
static constexpr int CS  = 1024;
static constexpr int CD  = 768;
static constexpr int CH  = 12;
static constexpr int CHD = 64;

typedef _Float16 f16;
typedef __attribute__((ext_vector_type(16))) _Float16 v16h;
typedef __attribute__((ext_vector_type(8)))  _Float16 v8h;
typedef __attribute__((ext_vector_type(8)))  float    v8f;

__device__ __forceinline__ v16h cat8(v8h lo, v8h hi) {
  return __builtin_shufflevector(lo, hi, 0,1,2,3,4,5,6,7,8,9,10,11,12,13,14,15);
}

// A-matrix fragment (16x32, f16). ISA layout: lane L holds row M=L%16.
// halves h=0..7  -> k = (L<16?0:8)  + h
// halves h=8..15 -> k = 16 + (L<16?0:8) + (h-8)
// => two contiguous 16B runs per lane.
__device__ __forceinline__ v16h ldA(const f16* base, int ld) {
  const int lane = threadIdx.x & 31;
  const f16* p = base + (size_t)(lane & 15) * ld + ((lane < 16) ? 0 : 8);
  v8h lo = *(const v8h*)(p);
  v8h hi = *(const v8h*)(p + 16);
  return cat8(lo, hi);
}

// B-matrix fragment (32x16, f16), stored with the K dimension contiguous per
// column n: lane L holds column N=L%16, halves h=0..7 -> k = (L<16?0:8)+h,
// halves h=8..15 -> k = 16 + (L<16?0:8) + (h-8).
// => two contiguous 16B runs per lane (two b128 loads).
__device__ __forceinline__ v16h ldB(const f16* base, int ld) {
  const int lane = threadIdx.x & 31;
  const f16* p = base + (size_t)(lane & 15) * ld + ((lane < 16) ? 0 : 8);
  v8h lo = *(const v8h*)(p);
  v8h hi = *(const v8h*)(p + 16);
  return cat8(lo, hi);
}

__device__ __forceinline__ v8f wmma_fp16(v16h a, v16h b, v8f c) {
  return __builtin_amdgcn_wmma_f32_16x16x32_f16(false, a, false, b,
                                                (short)0, c, false, false);
}

// Async DMA: memory -> LDS, 16 bytes per lane, tracked by ASYNCcnt.
// Inline asm keeps this portable across ROCm-7.2 / amdgpu-toolchain builtins.
__device__ __forceinline__ void async_copy_b128(void* lds_dst, const void* gsrc) {
  const unsigned loff = (unsigned)(uintptr_t)lds_dst;      // addr[31:0] = LDS offset
  const unsigned long long ga = (unsigned long long)(uintptr_t)gsrc;
  asm volatile("global_load_async_to_lds_b128 %0, %1, off"
               :: "v"(loff), "v"(ga) : "memory");
}
__device__ __forceinline__ void wait_asynccnt0() {
  asm volatile("s_wait_asynccnt 0" ::: "memory");
}

// ---------------------------------------------------------------------------
// f32 -> f16 conversion
// ---------------------------------------------------------------------------
__global__ void cvt_f32_to_f16(const float* __restrict__ in,
                               f16* __restrict__ out, int n) {
  int i = blockIdx.x * blockDim.x + threadIdx.x;
  if (i < n) out[i] = (f16)in[i];
}

// ---------------------------------------------------------------------------
// QKV GEMM: C[M=B*S, N=3D] = hs_f16 @ qkv_w^T + bias, then mask + scatter.
// Q,K laid out [(b*H+h)][S][HD]; V stored TRANSPOSED [(b*H+h)][HD][S] so the
// attention kernel can stage V^T tiles with pure contiguous async b128 copies.
// grid (M/64, 3D/128), block 128 (4 waves); each wave: 16 rows x 128 cols.
// ---------------------------------------------------------------------------
__global__ __launch_bounds__(128) void qkv_gemm_kernel(
    const f16* __restrict__ A, const f16* __restrict__ W,
    const float* __restrict__ bias, const unsigned char* __restrict__ mask,
    f16* __restrict__ Qd, f16* __restrict__ Kd, f16* __restrict__ VTd) {
  const int wave = threadIdx.x >> 5, lane = threadIdx.x & 31;
  const int m0 = blockIdx.x * 64 + wave * 16;
  const int n0 = blockIdx.y * 128;
  v8f acc[8] = {};
  const f16* Arow = A + (size_t)m0 * CD;
#pragma unroll
  for (int kk = 0; kk < CD; kk += 32) {
    if (kk + 64 < CD) __builtin_prefetch(Arow + kk + 64, 0, 3);
    v16h a = ldA(Arow + kk, CD);
    v16h bfr[8];
#pragma unroll
    for (int t = 0; t < 8; ++t)
      bfr[t] = ldB(W + (size_t)(n0 + t * 16) * CD + kk, CD);
#pragma unroll
    for (int t = 0; t < 8; ++t) acc[t] = wmma_fp16(a, bfr[t], acc[t]);
  }
  // epilogue: 768 % 128 == 0 so a strip never straddles the Q/K/V boundary
  const int sel = n0 / CD;
  const int b_  = m0 / CS;
  const int ro  = (lane < 16) ? 0 : 8;
#pragma unroll
  for (int t = 0; t < 8; ++t) {
    const int n  = n0 + t * 16 + (lane & 15);
    const int h  = (n % CD) / CHD;
    const int hd = n % CHD;
    const float bv = bias[n];
#pragma unroll
    for (int r = 0; r < 8; ++r) {
      const int m = m0 + r + ro;
      const int s = m - b_ * CS;
      const float val = acc[t][r] + bv;
      const f16 o = mask[(size_t)b_ * CS + s] ? (f16)val : (f16)0.0f;
      const size_t bh = (size_t)b_ * CH + h;
      if (sel == 0)      Qd[(bh * CS + s) * CHD + hd] = o;
      else if (sel == 1) Kd[(bh * CS + s) * CHD + hd] = o;
      else               VTd[(bh * CHD + hd) * CS + s] = o;
    }
  }
}

// ---------------------------------------------------------------------------
// Flash-style masked attention. grid = B*H*(S/128), block 256 (8 waves).
// Each wave: 16 query rows. Per 64-key block, K ([key][64]) and V^T ([hd][64])
// tiles are staged in LDS via GLOBAL_LOAD_ASYNC_TO_LDS_B128 (2 chunks per
// thread per tensor), shared by all 8 waves; 16 WMMAs between barriers.
// ---------------------------------------------------------------------------
__global__ __launch_bounds__(256) void attn_kernel(
    const f16* __restrict__ Qd, const f16* __restrict__ Kd,
    const f16* __restrict__ VTd, const unsigned char* __restrict__ mask,
    f16* __restrict__ ctx) {
  __shared__ f16 sK[64 * 64];    // [key][dim]   row-major
  __shared__ f16 sVT[64 * 64];   // [dim][key]
  __shared__ f16 sP[8][16 * 64]; // per-wave P staging (C-layout -> A-layout)

  const int tid = threadIdx.x, wave = tid >> 5, lane = tid & 31;
  const int nqb = CS / 128;
  const int bh = blockIdx.x / nqb;
  const int qb = blockIdx.x % nqb;
  const int b_ = bh / CH, h_ = bh % CH;
  const f16* Qp  = Qd  + (size_t)bh * CS * CHD;
  const f16* Kp  = Kd  + (size_t)bh * CS * CHD;
  const f16* VTp = VTd + (size_t)bh * CHD * CS;
  const unsigned char* mrow = mask + (size_t)b_ * CS;

  const int q0 = qb * 128 + wave * 16;
  const v16h qa0 = ldA(Qp + (size_t)q0 * CHD + 0, CHD);
  const v16h qa1 = ldA(Qp + (size_t)q0 * CHD + 32, CHD);

  v8f o[4] = {};
  float mrun[8], lrun[8];
#pragma unroll
  for (int r = 0; r < 8; ++r) { mrun[r] = -3.0e38f; lrun[r] = 0.f; }

  for (int k0 = 0; k0 < CS; k0 += 64) {
    __syncthreads();  // previous iteration done reading sK/sVT
    // K tile: 64*64 halves = 512 x 16B chunks (2 per thread), linear copy
#pragma unroll
    for (int c = tid; c < 512; c += 256)
      async_copy_b128(sK + c * 8, Kp + (size_t)k0 * CHD + c * 8);
    // V^T tile: 64 rows x 64 halves (8 chunks per row, 2 per thread)
#pragma unroll
    for (int c = tid; c < 512; c += 256) {
      const int hd = c >> 3, part = c & 7;
      async_copy_b128(sVT + hd * 64 + part * 8,
                      VTp + (size_t)hd * CS + k0 + part * 8);
    }
    wait_asynccnt0();
    __syncthreads();

    // scores: four 16x16 C-tiles over 64 keys, K-dim 64 = 2 WMMA steps each
    v8f s[4] = {};
#pragma unroll
    for (int t = 0; t < 4; ++t) {
      v16h bk0 = ldB(sK + t * 16 * 64 + 0, 64);
      v16h bk1 = ldB(sK + t * 16 * 64 + 32, 64);
      s[t] = wmma_fp16(qa0, bk0, s[t]);
      s[t] = wmma_fp16(qa1, bk1, s[t]);
    }

    // online softmax. C layout: lane = column (key), vgpr r = row (+8 hi half)
    bool kv[4];
#pragma unroll
    for (int t = 0; t < 4; ++t) kv[t] = mrow[k0 + t * 16 + (lane & 15)] != 0;
    float p[4][8], rm[8];
#pragma unroll
    for (int r = 0; r < 8; ++r) rm[r] = -3.0e38f;
#pragma unroll
    for (int t = 0; t < 4; ++t) {
      const float add = kv[t] ? 0.f : -3.0e38f;
#pragma unroll
      for (int r = 0; r < 8; ++r) {
        p[t][r] = s[t][r] * 0.125f + add;  // SCALE = HD^-0.5 = 1/8
        rm[r] = fmaxf(rm[r], p[t][r]);
      }
    }
#pragma unroll
    for (int off = 1; off < 16; off <<= 1)
#pragma unroll
      for (int r = 0; r < 8; ++r)
        rm[r] = fmaxf(rm[r], __shfl_xor(rm[r], off, 16));
    float corr[8], rs[8];
#pragma unroll
    for (int r = 0; r < 8; ++r) {
      const float mn = fmaxf(mrun[r], rm[r]);
      corr[r] = __expf(mrun[r] - mn);
      mrun[r] = mn;
      rs[r] = 0.f;
#pragma unroll
      for (int t = 0; t < 4; ++t) {
        p[t][r] = kv[t] ? __expf(p[t][r] - mn) : 0.f;  // exact 0 if masked
        rs[r] += p[t][r];
      }
    }
#pragma unroll
    for (int off = 1; off < 16; off <<= 1)
#pragma unroll
      for (int r = 0; r < 8; ++r)
        rs[r] += __shfl_xor(rs[r], off, 16);
#pragma unroll
    for (int r = 0; r < 8; ++r) {
      lrun[r] = lrun[r] * corr[r] + rs[r];
      o[0][r] *= corr[r]; o[1][r] *= corr[r];
      o[2][r] *= corr[r]; o[3][r] *= corr[r];
    }

    // P: C-layout regs -> LDS (row-major 16x64) -> two A-layout fragments
    {
      f16* pw = &sP[wave][0];
      const int ro = (lane < 16) ? 0 : 8;
#pragma unroll
      for (int r = 0; r < 8; ++r)
#pragma unroll
        for (int t = 0; t < 4; ++t)
          pw[(r + ro) * 64 + t * 16 + (lane & 15)] = (f16)p[t][r];
      asm volatile("s_wait_dscnt 0" ::: "memory");  // lane-crossing RAW in LDS
      v16h pa0 = ldA(pw + 0, 64);
      v16h pa1 = ldA(pw + 32, 64);
#pragma unroll
      for (int t = 0; t < 4; ++t) {
        v16h vb0 = ldB(sVT + t * 16 * 64 + 0, 64);
        v16h vb1 = ldB(sVT + t * 16 * 64 + 32, 64);
        o[t] = wmma_fp16(pa0, vb0, o[t]);
        o[t] = wmma_fp16(pa1, vb1, o[t]);
      }
    }
  }

  // finalize: divide by running sum (0 if row fully masked), store context
  // as [b][s][h*64+hd] f16 (i.e. transpose(0,2,1,3).reshape(B,S,D))
  const int ro = (lane < 16) ? 0 : 8;
  float inv[8];
#pragma unroll
  for (int r = 0; r < 8; ++r) inv[r] = (lrun[r] > 0.f) ? (1.f / lrun[r]) : 0.f;
#pragma unroll
  for (int r = 0; r < 8; ++r) {
    const int q = q0 + r + ro;
    f16* dst = ctx + ((size_t)(b_ * CS + q)) * CD + h_ * CHD;
#pragma unroll
    for (int t = 0; t < 4; ++t)
      dst[t * 16 + (lane & 15)] = (f16)(o[t][r] * inv[r]);
  }
}

// ---------------------------------------------------------------------------
// Output projection: out[M,768] = ctx @ proj_w^T + bias, masked rows zeroed.
// grid (M/64, 768/128), block 128; each wave: 16 rows x 128 cols.
// ---------------------------------------------------------------------------
__global__ __launch_bounds__(128) void proj_gemm_kernel(
    const f16* __restrict__ A, const f16* __restrict__ W,
    const float* __restrict__ bias, const unsigned char* __restrict__ mask,
    float* __restrict__ out) {
  const int wave = threadIdx.x >> 5, lane = threadIdx.x & 31;
  const int m0 = blockIdx.x * 64 + wave * 16;
  const int n0 = blockIdx.y * 128;
  v8f acc[8] = {};
  const f16* Arow = A + (size_t)m0 * CD;
#pragma unroll
  for (int kk = 0; kk < CD; kk += 32) {
    if (kk + 64 < CD) __builtin_prefetch(Arow + kk + 64, 0, 3);
    v16h a = ldA(Arow + kk, CD);
    v16h bfr[8];
#pragma unroll
    for (int t = 0; t < 8; ++t)
      bfr[t] = ldB(W + (size_t)(n0 + t * 16) * CD + kk, CD);
#pragma unroll
    for (int t = 0; t < 8; ++t) acc[t] = wmma_fp16(a, bfr[t], acc[t]);
  }
  const int ro = (lane < 16) ? 0 : 8;
#pragma unroll
  for (int t = 0; t < 8; ++t) {
    const int n = n0 + t * 16 + (lane & 15);
    const float bv = bias[n];
#pragma unroll
    for (int r = 0; r < 8; ++r) {
      const int m = m0 + r + ro;
      out[(size_t)m * CD + n] = mask[m] ? (acc[t][r] + bv) : 0.f;
    }
  }
}

// ---------------------------------------------------------------------------
extern "C" void kernel_launch(void* const* d_in, const int* in_sizes, int n_in,
                              void* d_out, int out_size, void* d_ws,
                              size_t ws_size, hipStream_t stream) {
  const float* hs            = (const float*)d_in[0];
  const unsigned char* mask  = (const unsigned char*)d_in[1];  // jax bool = 1B
  const float* qkvw          = (const float*)d_in[2];
  const float* qkvb          = (const float*)d_in[3];
  const float* projw         = (const float*)d_in[4];
  const float* projb         = (const float*)d_in[5];
  float* out                 = (float*)d_out;
  (void)in_sizes; (void)n_in; (void)out_size; (void)ws_size;

  char* ws = (char*)d_ws;
  size_t off = 0;
  auto take = [&](size_t bytes) -> char* {
    char* p = ws + off;
    off += (bytes + 255) & ~(size_t)255;
    return p;
  };

  f16* hs16 = (f16*)take((size_t)CB * CS * CD * sizeof(f16));
  f16* w1   = (f16*)take((size_t)3 * CD * CD * sizeof(f16));
  f16* w2   = (f16*)take((size_t)CD * CD * sizeof(f16));
  f16* Qd   = (f16*)take((size_t)CB * CH * CS * CHD * sizeof(f16));
  f16* Kd   = (f16*)take((size_t)CB * CH * CS * CHD * sizeof(f16));
  f16* VTd  = (f16*)take((size_t)CB * CH * CS * CHD * sizeof(f16));
  f16* ctx  = (f16*)take((size_t)CB * CS * CD * sizeof(f16));

  const int n1 = CB * CS * CD;
  const int n2 = 3 * CD * CD;
  const int n3 = CD * CD;
  cvt_f32_to_f16<<<(n1 + 255) / 256, 256, 0, stream>>>(hs, hs16, n1);
  cvt_f32_to_f16<<<(n2 + 255) / 256, 256, 0, stream>>>(qkvw, w1, n2);
  cvt_f32_to_f16<<<(n3 + 255) / 256, 256, 0, stream>>>(projw, w2, n3);

  qkv_gemm_kernel<<<dim3((CB * CS) / 64, (3 * CD) / 128), 128, 0, stream>>>(
      hs16, w1, qkvb, mask, Qd, Kd, VTd);

  attn_kernel<<<dim3(CB * CH * (CS / 128)), 256, 0, stream>>>(
      Qd, Kd, VTd, mask, ctx);

  proj_gemm_kernel<<<dim3((CB * CS) / 64, CD / 128), 128, 0, stream>>>(
      ctx, w2, projb, mask, out);
}
